// IT_Fast_Attn_76656576299169
// MI455X (gfx1250) — matrix operand order repaired and checked by reading, not compiled, not run on hardware
//
#include <hip/hip_runtime.h>
#include <hip/hip_bf16.h>

// MI455X / gfx1250 implementation. All GEMMs via v_wmma_f32_16x16x32_f16
// (f16 operands, f32 accumulation); softmax/LayerNorm in fp32 VALU with
// wave32 shuffle reductions. k_ctx stages tiles via async global->LDS copies
// (CDNA5 GLOBAL_LOAD_ASYNC_TO_LDS_B128 + s_wait_asynccnt).
// Workspace requirement: ~137 MB.

typedef __attribute__((ext_vector_type(16))) _Float16 v16h;
typedef __attribute__((ext_vector_type(8)))  float    v8f;

#define B_   8
#define C_   256
#define N_   4096
#define SZ_  (B_ * N_ * C_)      // 8388608 elements
#define LN_EPS 1e-5f

static __device__ __forceinline__ int lane_id() { return threadIdx.x & 31; }
static __device__ __forceinline__ int wave_id() { return threadIdx.x >> 5; }

// wave32 WMMA fragment coordinate maps (CDNA5 ISA 7.12.2)
// A: 16x32 f16. lane L holds row M=L&15; half h -> K.
static __device__ __forceinline__ void a_coord(int lane, int h, int& m, int& k) {
    int j = h >> 1, p = h & 1;
    m = lane & 15;
    k = ((j >> 2) << 4) | (((lane >> 4) & 1) << 3) | ((j & 3) << 1) | p;
}
// B: 32x16 f16. lane L holds col N=L&15; half h -> K.
static __device__ __forceinline__ void b_coord(int lane, int h, int& k, int& n) {
    n = lane & 15;
    k = ((h >> 3) << 4) | (((lane >> 4) & 1) << 3) | (h & 7);
}
// C/D: 16x16 f32, 8 VGPRs.
static __device__ __forceinline__ void cd_coord(int lane, int r, int& m, int& n) {
    m = r | (((lane >> 4) & 1) << 3);
    n = lane & 15;
}

static __device__ __forceinline__ v8f wmma_f16(v16h a, v16h b, v8f c) {
    return __builtin_amdgcn_wmma_f32_16x16x32_f16(false, a, false, b,
                                                  (short)0, c, false, false);
}

static __device__ __forceinline__ float wave_sum(float x) {
#pragma unroll
    for (int off = 16; off; off >>= 1) x += __shfl_xor(x, off, 32);
    return x;
}
static __device__ __forceinline__ float wave_max(float x) {
#pragma unroll
    for (int off = 16; off; off >>= 1) x = fmaxf(x, __shfl_xor(x, off, 32));
    return x;
}

// ---------------------------------------------------------------- K0: weights -> f16
__global__ void k_cvt_weights(const float* __restrict__ Wq, const float* __restrict__ Wm,
                              const float* __restrict__ Wr, _Float16* __restrict__ Wq_h,
                              _Float16* __restrict__ Wm_h, _Float16* __restrict__ Wr_h) {
    int i = blockIdx.x * blockDim.x + threadIdx.x;
    if (i < C_ * C_) Wq_h[i] = (_Float16)Wq[i];
    if (i < 4 * C_ * C_) { Wm_h[i] = (_Float16)Wm[i]; Wr_h[i] = (_Float16)Wr[i]; }
}

// ------------------------------------------------- K1: xw = x^T @ Wqkv^T  (fused transpose)
// x: (B,C,N) f32.  xw: (B,N,C).  M=n rows, K=c, 16x16 tile per wave, 2 col-tiles/wave.
__global__ void k_qkv(const float* __restrict__ x, const _Float16* __restrict__ Wq_h,
                      float* __restrict__ xw_f32, _Float16* __restrict__ xw_h,
                      _Float16* __restrict__ v_h) {
    const int b = blockIdx.y, n0 = blockIdx.x * 16;
    const int lane = lane_id(), w = wave_id();
    const float* xb = x + (size_t)b * C_ * N_;
    v8f acc0 = {}, acc1 = {};
    for (int ks = 0; ks < C_; ks += 32) {
        v16h a;
#pragma unroll
        for (int h = 0; h < 16; ++h) {
            int m, k; a_coord(lane, h, m, k);
            a[h] = (_Float16)xb[(size_t)(ks + k) * N_ + (n0 + m)];
        }
#pragma unroll
        for (int t = 0; t < 2; ++t) {
            int c0 = (w + t * 8) * 16;
            v16h bf;
#pragma unroll
            for (int h = 0; h < 16; ++h) {
                int k, n; b_coord(lane, h, k, n);
                bf[h] = Wq_h[(size_t)(c0 + n) * C_ + (ks + k)];
            }
            if (t == 0) acc0 = wmma_f16(a, bf, acc0);
            else        acc1 = wmma_f16(a, bf, acc1);
        }
    }
#pragma unroll
    for (int t = 0; t < 2; ++t) {
        int c0 = (w + t * 8) * 16;
        v8f acc = t ? acc1 : acc0;
#pragma unroll
        for (int r = 0; r < 8; ++r) {
            int m, n; cd_coord(lane, r, m, n);
            size_t idx = ((size_t)b * N_ + (n0 + m)) * C_ + (c0 + n);
            float vv = acc[r];
            xw_f32[idx] = vv;
            _Float16 hv = (_Float16)vv;
            xw_h[idx] = hv;
            v_h[idx]  = hv;   // v starts as a copy of x
        }
    }
}

// ---------------------------------------------------------------- K2: softmax over C (once)
__global__ void k_softmax(const float* __restrict__ xw, _Float16* __restrict__ sm) {
    const int lane = lane_id(), w = wave_id();
    const size_t row = (size_t)blockIdx.x * 8 + w;   // over B*N rows
    const float* p = xw + row * C_;
    float vals[8], mx = -1e30f;
#pragma unroll
    for (int q = 0; q < 8; ++q) { vals[q] = p[lane + 32 * q]; mx = fmaxf(mx, vals[q]); }
    mx = wave_max(mx);
    float s = 0.f;
#pragma unroll
    for (int q = 0; q < 8; ++q) { vals[q] = __expf(vals[q] - mx); s += vals[q]; }
    s = wave_sum(s);
    float inv = 1.0f / s;
    _Float16* o = sm + row * C_;
#pragma unroll
    for (int q = 0; q < 8; ++q) o[lane + 32 * q] = (_Float16)(vals[q] * inv);
}

// ------------------------------------------- K3: ctx[c,d] = sum_n x[n,c] v[n,d]  (split-K)
// Per wave: stage 32x16 x/v slabs into LDS via async global->LDS b128 copies
// (coalesced rows), then gather the K-strided WMMA fragments from LDS.
__global__ void k_ctx(const _Float16* __restrict__ x_h, const _Float16* __restrict__ v_h,
                      _Float16* __restrict__ ctx_h) {
    __shared__ float red[8][256];                                   // 8 KB
    __shared__ __attribute__((aligned(16))) _Float16 stage[8][2][32 * 16]; // 16 KB
    const int b = blockIdx.z;
    const int c0 = blockIdx.x * 16, d0 = blockIdx.y * 16;
    const int lane = lane_id(), w = wave_id();
    const _Float16* xb = x_h + (size_t)b * N_ * C_ + c0;
    const _Float16* vb = v_h + (size_t)b * N_ * C_ + d0;
    _Float16* sx = &stage[w][0][0];
    _Float16* sv = &stage[w][1][0];
    // LDS byte offsets for this lane's staged row (generic-address low 32 bits)
    const unsigned lx = (unsigned)(uintptr_t)sx + (unsigned)lane * 32u;
    const unsigned lv = (unsigned)(uintptr_t)sv + (unsigned)lane * 32u;
    v8f acc = {};
    for (int i = 0; i < 16; ++i) {             // 8 waves x 16 steps x 32 = K=4096
        int ks = (w * 16 + i) * 32;
        // lane L copies global row (ks+L) x 16 halves (32 B) of each operand
        unsigned long long gx = (unsigned long long)(uintptr_t)(xb + (size_t)(ks + lane) * C_);
        unsigned long long gv = (unsigned long long)(uintptr_t)(vb + (size_t)(ks + lane) * C_);
        asm volatile(
            "global_load_async_to_lds_b128 %0, %1, off\n\t"
            "global_load_async_to_lds_b128 %2, %3, off\n\t"
            "global_load_async_to_lds_b128 %4, %5, off\n\t"
            "global_load_async_to_lds_b128 %6, %7, off\n\t"
            "s_wait_asynccnt 0x0"
            :: "v"(lx), "v"(gx), "v"(lx + 16u), "v"(gx + 16ull),
               "v"(lv), "v"(gv), "v"(lv + 16u), "v"(gv + 16ull)
            : "memory");
        v16h a, bf;
#pragma unroll
        for (int h = 0; h < 16; ++h) {
            int m, k; a_coord(lane, h, m, k);
            a[h] = sx[k * 16 + m];             // slab is (k-row, col) row-major
            int kk, n; b_coord(lane, h, kk, n);
            bf[h] = sv[kk * 16 + n];
        }
        acc = wmma_f16(a, bf, acc);
        // next iteration's async copies issue only after the s_wait_dscnt that
        // precedes the WMMA consuming a/bf, so the slab is safe to overwrite.
    }
#pragma unroll
    for (int r = 0; r < 8; ++r) red[w][r * 32 + lane] = acc[r];
    __syncthreads();
    {
        int r = threadIdx.x >> 5, l = threadIdx.x & 31;
        float s = 0.f;
#pragma unroll
        for (int ww = 0; ww < 8; ++ww) s += red[ww][r * 32 + l];
        int m, n; cd_coord(l, r, m, n);
        ctx_h[((size_t)b * C_ + (c0 + m)) * C_ + (d0 + n)] = (_Float16)s;
    }
}

// ---------------------------------------------------------------- K4: out = softmax(x) @ ctx
__global__ void k_out(const _Float16* __restrict__ sm, const _Float16* __restrict__ ctx_h,
                      _Float16* __restrict__ out_h) {
    const int b = blockIdx.y, n0 = blockIdx.x * 16;
    const int lane = lane_id(), w = wave_id();
    const _Float16* smb = sm + (size_t)b * N_ * C_;
    const _Float16* cb  = ctx_h + (size_t)b * C_ * C_;
    v8f acc0 = {}, acc1 = {};
    for (int ks = 0; ks < C_; ks += 32) {
        v16h a;
#pragma unroll
        for (int h = 0; h < 16; ++h) {
            int m, k; a_coord(lane, h, m, k);
            a[h] = smb[(size_t)(n0 + m) * C_ + (ks + k)];
        }
#pragma unroll
        for (int t = 0; t < 2; ++t) {
            int d0 = (w + t * 8) * 16;
            v16h bf;
#pragma unroll
            for (int h = 0; h < 16; ++h) {
                int k, n; b_coord(lane, h, k, n);
                bf[h] = cb[(size_t)(ks + k) * C_ + (d0 + n)];
            }
            if (t == 0) acc0 = wmma_f16(a, bf, acc0);
            else        acc1 = wmma_f16(a, bf, acc1);
        }
    }
#pragma unroll
    for (int t = 0; t < 2; ++t) {
        int d0 = (w + t * 8) * 16;
        v8f acc = t ? acc1 : acc0;
#pragma unroll
        for (int r = 0; r < 8; ++r) {
            int m, n; cd_coord(lane, r, m, n);
            out_h[((size_t)b * N_ + (n0 + m)) * C_ + (d0 + n)] = (_Float16)acc[r];
        }
    }
}

// ------------------------------- K5: mlp = relu([out,v] @ Wmlp^T + b)  (concat via pointer)
__global__ void k_mlp(const _Float16* __restrict__ out_h, const _Float16* __restrict__ v_h,
                      const _Float16* __restrict__ Wm_h, const float* __restrict__ bias,
                      _Float16* __restrict__ mlp_h) {
    const int b = blockIdx.y, n0 = blockIdx.x * 16;
    const int lane = lane_id(), w = wave_id();
    v8f acc[4] = {};
    for (int ks = 0; ks < 2 * C_; ks += 32) {
        const _Float16* src = (ks < C_) ? out_h : v_h;
        const int kb = (ks < C_) ? ks : ks - C_;
        v16h a;
#pragma unroll
        for (int h = 0; h < 16; ++h) {
            int m, k; a_coord(lane, h, m, k);
            a[h] = src[((size_t)b * N_ + (n0 + m)) * C_ + (kb + k)];
        }
#pragma unroll
        for (int t = 0; t < 4; ++t) {
            int c0 = (w + t * 8) * 16;
            v16h bf;
#pragma unroll
            for (int h = 0; h < 16; ++h) {
                int k, n; b_coord(lane, h, k, n);
                bf[h] = Wm_h[(size_t)(c0 + n) * (2 * C_) + (ks + k)];
            }
            acc[t] = wmma_f16(a, bf, acc[t]);
        }
    }
#pragma unroll
    for (int t = 0; t < 4; ++t) {
        int c0 = (w + t * 8) * 16;
#pragma unroll
        for (int r = 0; r < 8; ++r) {
            int m, n; cd_coord(lane, r, m, n);
            float vv = fmaxf(acc[t][r] + bias[c0 + n], 0.f);
            mlp_h[((size_t)b * N_ + (n0 + m)) * (2 * C_) + (c0 + n)] = (_Float16)vv;
        }
    }
}

// ---------------------------------------- K6: LN1 stats (mean, rstd) over cat2 = [out,v,mlp]
__global__ void k_ln1_stats(const _Float16* __restrict__ out_h, const _Float16* __restrict__ v_h,
                            const _Float16* __restrict__ mlp_h, float* __restrict__ stats) {
    const int lane = lane_id(), w = wave_id();
    const size_t row = (size_t)blockIdx.x * 8 + w;
    const _Float16* po = out_h + row * C_;
    const _Float16* pv = v_h   + row * C_;
    const _Float16* pm = mlp_h + row * (2 * C_);
    float s = 0.f, ss = 0.f;
#pragma unroll
    for (int q = 0; q < 8; ++q)  { float f = (float)po[lane + 32 * q]; s += f; ss += f * f; }
#pragma unroll
    for (int q = 0; q < 8; ++q)  { float f = (float)pv[lane + 32 * q]; s += f; ss += f * f; }
#pragma unroll
    for (int q = 0; q < 16; ++q) { float f = (float)pm[lane + 32 * q]; s += f; ss += f * f; }
    s = wave_sum(s); ss = wave_sum(ss);
    if (lane == 0) {
        float mu  = s * (1.0f / (4 * C_));
        float var = ss * (1.0f / (4 * C_)) - mu * mu;
        stats[row * 2 + 0] = mu;
        stats[row * 2 + 1] = rsqrtf(var + LN_EPS);
    }
}

// --------- K7: v = LN2( 2x + LN1(cat2)@Wres^T ); LN1 fused into A-fragment load, LN2 via LDS
__global__ void k_res(const _Float16* __restrict__ out_h, const _Float16* __restrict__ v_h,
                      const _Float16* __restrict__ mlp_h, const float* __restrict__ stats,
                      const _Float16* __restrict__ Wr_h, const float* __restrict__ g1,
                      const float* __restrict__ beta1, const float* __restrict__ xw_f32,
                      const float* __restrict__ g2, const float* __restrict__ beta2,
                      _Float16* __restrict__ vout_h, float* __restrict__ final_out, int last) {
    __shared__ float tile[16 * C_];   // 16 rows x 256 cols, 16 KB
    const int b = blockIdx.y, n0 = blockIdx.x * 16;
    const int lane = lane_id(), w = wave_id();
    const int myrow = n0 + (lane & 15);               // A fragment row M = lane&15
    const float mu   = stats[((size_t)b * N_ + myrow) * 2 + 0];
    const float rstd = stats[((size_t)b * N_ + myrow) * 2 + 1];
    v8f acc0 = {}, acc1 = {};
    for (int ks = 0; ks < 4 * C_; ks += 32) {
        const _Float16* src; int kb, stride;
        if (ks < C_)        { src = out_h; kb = ks;          stride = C_; }
        else if (ks < 2*C_) { src = v_h;   kb = ks - C_;     stride = C_; }
        else                { src = mlp_h; kb = ks - 2 * C_; stride = 2 * C_; }
        v16h a;
#pragma unroll
        for (int h = 0; h < 16; ++h) {
            int m, k; a_coord(lane, h, m, k);
            float raw = (float)src[((size_t)b * N_ + (n0 + m)) * stride + (kb + k)];
            a[h] = (_Float16)((raw - mu) * rstd * g1[ks + k] + beta1[ks + k]);
        }
#pragma unroll
        for (int t = 0; t < 2; ++t) {
            int c0 = (w + t * 8) * 16;
            v16h bf;
#pragma unroll
            for (int h = 0; h < 16; ++h) {
                int k, n; b_coord(lane, h, k, n);
                bf[h] = Wr_h[(size_t)(c0 + n) * (4 * C_) + (ks + k)];
            }
            if (t == 0) acc0 = wmma_f16(a, bf, acc0);
            else        acc1 = wmma_f16(a, bf, acc1);
        }
    }
    // t = x + (GEMM + x) = 2x + GEMM
#pragma unroll
    for (int t = 0; t < 2; ++t) {
        int c0 = (w + t * 8) * 16;
        v8f acc = t ? acc1 : acc0;
#pragma unroll
        for (int r = 0; r < 8; ++r) {
            int m, n; cd_coord(lane, r, m, n);
            float xv = xw_f32[((size_t)b * N_ + (n0 + m)) * C_ + (c0 + n)];
            tile[m * C_ + (c0 + n)] = acc[r] + 2.0f * xv;
        }
    }
    __syncthreads();
    // LN2 over C per row; each wave owns 2 rows
#pragma unroll
    for (int rr = 0; rr < 2; ++rr) {
        int m = w * 2 + rr;
        float vals[8], s = 0.f, ss = 0.f;
#pragma unroll
        for (int q = 0; q < 8; ++q) {
            vals[q] = tile[m * C_ + lane + 32 * q];
            s += vals[q]; ss += vals[q] * vals[q];
        }
        s = wave_sum(s); ss = wave_sum(ss);
        float mu2   = s * (1.0f / C_);
        float rstd2 = rsqrtf(ss * (1.0f / C_) - mu2 * mu2 + LN_EPS);
#pragma unroll
        for (int q = 0; q < 8; ++q) {
            int c = lane + 32 * q;
            float vv = (vals[q] - mu2) * rstd2 * g2[c] + beta2[c];
            if (last) final_out[((size_t)b * C_ + c) * N_ + (n0 + m)] = vv; // (B,C,N)
            else      vout_h[((size_t)b * N_ + (n0 + m)) * C_ + c] = (_Float16)vv;
        }
    }
}

extern "C" void kernel_launch(void* const* d_in, const int* in_sizes, int n_in,
                              void* d_out, int out_size, void* d_ws, size_t ws_size,
                              hipStream_t stream) {
    const float* x     = (const float*)d_in[0];
    const float* W_qkv = (const float*)d_in[1];
    const float* W_mlp = (const float*)d_in[2];
    const float* b_mlp = (const float*)d_in[3];
    const float* g1    = (const float*)d_in[4];
    const float* beta1 = (const float*)d_in[5];
    const float* W_res = (const float*)d_in[6];
    const float* g2    = (const float*)d_in[7];
    const float* beta2 = (const float*)d_in[8];
    float* out = (float*)d_out;

    char* p = (char*)d_ws;
    auto alloc = [&](size_t bytes) {
        char* r = p; p += (bytes + 255) & ~(size_t)255; return (void*)r;
    };
    float*    xw_f32 = (float*)   alloc((size_t)SZ_ * 4);
    float*    stats  = (float*)   alloc((size_t)B_ * N_ * 2 * 4);
    _Float16* xw_h   = (_Float16*)alloc((size_t)SZ_ * 2);
    _Float16* sm_h   = (_Float16*)alloc((size_t)SZ_ * 2);
    _Float16* v_h    = (_Float16*)alloc((size_t)SZ_ * 2);
    _Float16* out_h  = (_Float16*)alloc((size_t)SZ_ * 2);
    _Float16* mlp_h  = (_Float16*)alloc((size_t)SZ_ * 4);
    _Float16* ctx_h  = (_Float16*)alloc((size_t)B_ * C_ * C_ * 2);
    _Float16* Wq_h   = (_Float16*)alloc((size_t)C_ * C_ * 2);
    _Float16* Wm_h   = (_Float16*)alloc((size_t)4 * C_ * C_ * 2);
    _Float16* Wr_h   = (_Float16*)alloc((size_t)4 * C_ * C_ * 2);

    dim3 blk(256);  // 8 wave32 waves per block
    k_cvt_weights<<<dim3((4 * C_ * C_ + 255) / 256), blk, 0, stream>>>(
        W_qkv, W_mlp, W_res, Wq_h, Wm_h, Wr_h);
    k_qkv<<<dim3(N_ / 16, B_), blk, 0, stream>>>(x, Wq_h, xw_f32, xw_h, v_h);
    k_softmax<<<dim3(B_ * N_ / 8), blk, 0, stream>>>(xw_f32, sm_h);
    for (int it = 0; it < 2; ++it) {
        k_ctx<<<dim3(C_ / 16, C_ / 16, B_), blk, 0, stream>>>(xw_h, v_h, ctx_h);
        k_out<<<dim3(N_ / 16, B_), blk, 0, stream>>>(sm_h, ctx_h, out_h);
        k_mlp<<<dim3(N_ / 16, B_), blk, 0, stream>>>(out_h, v_h, Wm_h, b_mlp, mlp_h);
        k_ln1_stats<<<dim3(B_ * N_ / 8), blk, 0, stream>>>(out_h, v_h, mlp_h, stats);
        k_res<<<dim3(N_ / 16, B_), blk, 0, stream>>>(out_h, v_h, mlp_h, stats, Wr_h,
                                                     g1, beta1, xw_f32, g2, beta2,
                                                     v_h, out, it == 1);
    }
    (void)in_sizes; (void)n_in; (void)out_size; (void)ws_size;
}
